// OnlineClustering_57913339019525
// MI455X (gfx1250) — compile-verified, matrix operand order checked by reading, not run on hardware
//
#include <hip/hip_runtime.h>
#include <hip/hip_bf16.h>
#include <stdint.h>

// ---------------- problem constants ----------------
#define IN_DIM   768
#define OUT_DIM  4096
#define NBATCH   32
#define NTOK     1024
#define NROWS    (NBATCH * NTOK)          // 32768
#define INV_TT   (1.0f / 0.06f)
#define INV_PT   (1.0f / 0.12f)
#define EXP_MAX  50.0f
#define EPS_NORM 1e-7f
#define EPS_SK   1e-8f

// ---------------- WMMA types ----------------
typedef __attribute__((ext_vector_type(16))) __bf16 bf16x16;
typedef __attribute__((ext_vector_type(8)))  float  f32x8;

struct Bytes32 { uint4 lo; uint4 hi; };   // 32B container for a bf16x16 fragment

__device__ __forceinline__ bf16x16 frag_from(const uint4* __restrict__ row, int qlo, int qhi) {
    Bytes32 b;
    b.lo = row[qlo];
    b.hi = row[qhi];
    return __builtin_bit_cast(bf16x16, b);
}

// float -> bf16 bits, round-to-nearest-even
__device__ __forceinline__ uint16_t f2bf(float f) {
    uint32_t u = __float_as_uint(f);
    u += 0x7fffu + ((u >> 16) & 1u);
    return (uint16_t)(u >> 16);
}

// 8 WMMAs of a 32x64 wave tile
__device__ __forceinline__ void wmma8(const bf16x16& a0, const bf16x16& a1,
                                      const bf16x16& b0, const bf16x16& b1,
                                      const bf16x16& b2, const bf16x16& b3,
                                      f32x8& acc00, f32x8& acc01, f32x8& acc02, f32x8& acc03,
                                      f32x8& acc10, f32x8& acc11, f32x8& acc12, f32x8& acc13) {
    acc00 = __builtin_amdgcn_wmma_f32_16x16x32_bf16(false, a0, false, b0, (short)0, acc00, false, false);
    acc01 = __builtin_amdgcn_wmma_f32_16x16x32_bf16(false, a0, false, b1, (short)0, acc01, false, false);
    acc02 = __builtin_amdgcn_wmma_f32_16x16x32_bf16(false, a0, false, b2, (short)0, acc02, false, false);
    acc03 = __builtin_amdgcn_wmma_f32_16x16x32_bf16(false, a0, false, b3, (short)0, acc03, false, false);
    acc10 = __builtin_amdgcn_wmma_f32_16x16x32_bf16(false, a1, false, b0, (short)0, acc10, false, false);
    acc11 = __builtin_amdgcn_wmma_f32_16x16x32_bf16(false, a1, false, b1, (short)0, acc11, false, false);
    acc12 = __builtin_amdgcn_wmma_f32_16x16x32_bf16(false, a1, false, b2, (short)0, acc12, false, false);
    acc13 = __builtin_amdgcn_wmma_f32_16x16x32_bf16(false, a1, false, b3, (short)0, acc13, false, false);
}

// ---------------- fixed-order block reductions (blockDim.x == 256) ----------------
__device__ __forceinline__ float block_sum(float v, float* red) {
    int t = threadIdx.x;
    red[t] = v; __syncthreads();
    for (int s = 128; s > 0; s >>= 1) { if (t < s) red[t] += red[t + s]; __syncthreads(); }
    float r = red[0]; __syncthreads();
    return r;
}
__device__ __forceinline__ float block_max(float v, float* red) {
    int t = threadIdx.x;
    red[t] = v; __syncthreads();
    for (int s = 128; s > 0; s >>= 1) { if (t < s) red[t] = fmaxf(red[t], red[t + s]); __syncthreads(); }
    float r = red[0]; __syncthreads();
    return r;
}

// ================= kernel 1: row L2-normalize x and convert to bf16 =================
__global__ void k_rownorm_cvt(const float* __restrict__ x, uint16_t* __restrict__ xb) {
    __shared__ float red[256];
    const int row = blockIdx.x;
    const size_t base = (size_t)row * IN_DIM;
    const int t = threadIdx.x;
    float v0 = x[base + t];
    float v1 = x[base + t + 256];
    float v2 = x[base + t + 512];
    float ss = v0 * v0 + v1 * v1 + v2 * v2;
    float s = block_sum(ss, red);
    float inv = 1.0f / fmaxf(sqrtf(s), EPS_NORM);
    xb[base + t]       = f2bf(v0 * inv);
    xb[base + t + 256] = f2bf(v1 * inv);
    xb[base + t + 512] = f2bf(v2 * inv);
}

// ================= kernel 2: convert W to bf16 =================
__global__ void k_cvt_w(const float* __restrict__ w, uint16_t* __restrict__ wb) {
    size_t i = (size_t)blockIdx.x * blockDim.x + threadIdx.x;
    if (i < (size_t)OUT_DIM * IN_DIM) wb[i] = f2bf(w[i]);
}

// ================= kernel 3: WMMA GEMM  logits[32768, 4096] = Xn * W^T =================
// block = 128 threads (4 waves). Wave tile: 32 (M) x 64 (N) = 8 WMMA tiles.
// B fragments ping-pong double-buffered across K (no register rotation copies);
// A fragments loaded per-stage. grid = (OUT_DIM/256, NROWS/32) = (16, 1024)
__global__ void __launch_bounds__(128, 1)
k_gemm_wmma(const uint16_t* __restrict__ xb,
            const uint16_t* __restrict__ wb,
            float* __restrict__ logits) {
    const int wave = threadIdx.x >> 5;
    const int lane = threadIdx.x & 31;
    const int l16  = lane & 15;
    const int hi   = lane >> 4;            // 0: K chunks {0..7,16..23}, 1: {8..15,24..31}

    const int m0 = blockIdx.y * 32;
    const int n0 = blockIdx.x * 256 + wave * 64;

    const uint4* rowA0 = (const uint4*)(xb + (size_t)(m0 +  0 + l16) * IN_DIM);
    const uint4* rowA1 = (const uint4*)(xb + (size_t)(m0 + 16 + l16) * IN_DIM);
    const uint4* rowB0 = (const uint4*)(wb + (size_t)(n0 +  0 + l16) * IN_DIM);
    const uint4* rowB1 = (const uint4*)(wb + (size_t)(n0 + 16 + l16) * IN_DIM);
    const uint4* rowB2 = (const uint4*)(wb + (size_t)(n0 + 32 + l16) * IN_DIM);
    const uint4* rowB3 = (const uint4*)(wb + (size_t)(n0 + 48 + l16) * IN_DIM);

    f32x8 acc00 = {}, acc01 = {}, acc02 = {}, acc03 = {};
    f32x8 acc10 = {}, acc11 = {}, acc12 = {}, acc13 = {};

    // ---- prologue: buffer X <- B fragments for K=0 ----
    bf16x16 bx0 = frag_from(rowB0, hi, 2 + hi);
    bf16x16 bx1 = frag_from(rowB1, hi, 2 + hi);
    bf16x16 bx2 = frag_from(rowB2, hi, 2 + hi);
    bf16x16 bx3 = frag_from(rowB3, hi, 2 + hi);
    bf16x16 by0, by1, by2, by3;

    // K-steps: 24 total (step = 32). Loop covers 22 (11 iters x 2 stages), tail covers 2.
#pragma unroll 1
    for (int k = 0; k < IN_DIM - 64; k += 64) {
        const int kq = k >> 3;

        __builtin_prefetch(rowA0 + kq + 12, 0, 1);
        __builtin_prefetch(rowB0 + kq + 12, 0, 1);

        // ---- stage 1: compute with X (K=k), load Y (K=k+32) ----
        {
            bf16x16 a0 = frag_from(rowA0, kq + hi, kq + 2 + hi);
            bf16x16 a1 = frag_from(rowA1, kq + hi, kq + 2 + hi);
            by0 = frag_from(rowB0, kq + 4 + hi, kq + 6 + hi);
            by1 = frag_from(rowB1, kq + 4 + hi, kq + 6 + hi);
            by2 = frag_from(rowB2, kq + 4 + hi, kq + 6 + hi);
            by3 = frag_from(rowB3, kq + 4 + hi, kq + 6 + hi);
            wmma8(a0, a1, bx0, bx1, bx2, bx3,
                  acc00, acc01, acc02, acc03, acc10, acc11, acc12, acc13);
        }
        // ---- stage 2: compute with Y (K=k+32), load X (K=k+64) ----
        {
            const int kq2 = kq + 4;
            bf16x16 a0 = frag_from(rowA0, kq2 + hi, kq2 + 2 + hi);
            bf16x16 a1 = frag_from(rowA1, kq2 + hi, kq2 + 2 + hi);
            bx0 = frag_from(rowB0, kq2 + 4 + hi, kq2 + 6 + hi);
            bx1 = frag_from(rowB1, kq2 + 4 + hi, kq2 + 6 + hi);
            bx2 = frag_from(rowB2, kq2 + 4 + hi, kq2 + 6 + hi);
            bx3 = frag_from(rowB3, kq2 + 4 + hi, kq2 + 6 + hi);
            wmma8(a0, a1, by0, by1, by2, by3,
                  acc00, acc01, acc02, acc03, acc10, acc11, acc12, acc13);
        }
    }

    // ---- tail: K = IN_DIM-64 (X resident), then K = IN_DIM-32 (Y) ----
    {
        const int kq = (IN_DIM - 64) >> 3;
        bf16x16 a0 = frag_from(rowA0, kq + hi, kq + 2 + hi);
        bf16x16 a1 = frag_from(rowA1, kq + hi, kq + 2 + hi);
        by0 = frag_from(rowB0, kq + 4 + hi, kq + 6 + hi);
        by1 = frag_from(rowB1, kq + 4 + hi, kq + 6 + hi);
        by2 = frag_from(rowB2, kq + 4 + hi, kq + 6 + hi);
        by3 = frag_from(rowB3, kq + 4 + hi, kq + 6 + hi);
        wmma8(a0, a1, bx0, bx1, bx2, bx3,
              acc00, acc01, acc02, acc03, acc10, acc11, acc12, acc13);
    }
    {
        const int kq = (IN_DIM - 32) >> 3;
        bf16x16 a0 = frag_from(rowA0, kq + hi, kq + 2 + hi);
        bf16x16 a1 = frag_from(rowA1, kq + hi, kq + 2 + hi);
        wmma8(a0, a1, by0, by1, by2, by3,
              acc00, acc01, acc02, acc03, acc10, acc11, acc12, acc13);
    }

    // C/D layout: lanes 0-15 -> N=lane, M = 8*hi + vgpr_index
#pragma unroll
    for (int i = 0; i < 8; ++i) {
        const int mrow = hi * 8 + i;
        float* dst0 = logits + (size_t)(m0 + mrow)      * OUT_DIM + n0 + l16;
        float* dst1 = logits + (size_t)(m0 + 16 + mrow) * OUT_DIM + n0 + l16;
        dst0[0]  = acc00[i];
        dst0[16] = acc01[i];
        dst0[32] = acc02[i];
        dst0[48] = acc03[i];
        dst1[0]  = acc10[i];
        dst1[16] = acc11[i];
        dst1[32] = acc12[i];
        dst1[48] = acc13[i];
    }
}

// ================= kernel 4: init Sinkhorn scale vectors =================
__global__ void k_init_rc(float* __restrict__ r, float* __restrict__ c) {
    int i = blockIdx.x * blockDim.x + threadIdx.x;
    if (i < NBATCH * OUT_DIM) c[i] = 1.0f;
    if (i < NROWS) r[i] = 1.0f;
}

// ================= kernel 5: column max over n (per batch) =================
__global__ void k_colmax(const float* __restrict__ logits, float* __restrict__ colmax) {
    const int p = blockIdx.x * 256 + threadIdx.x;
    const int b = blockIdx.y;
    const float* col = logits + ((size_t)b * NTOK) * OUT_DIM + p;
    float mx = -3.402823466e+38f;
    for (int n = 0; n < NTOK; ++n) mx = fmaxf(mx, col[(size_t)n * OUT_DIM]);
    colmax[b * OUT_DIM + p] = mx;
}

// ================= kernel 6: SK column pass =================
__global__ void k_sk_col(const float* __restrict__ logits,
                         const float* __restrict__ colmax,
                         const float* __restrict__ r,
                         float* __restrict__ c) {
    const int p = blockIdx.x * 256 + threadIdx.x;
    const int b = blockIdx.y;
    const float mx = colmax[b * OUT_DIM + p];
    const float* col = logits + ((size_t)b * NTOK) * OUT_DIM + p;
    const float* rb  = r + b * NTOK;
    float t = 0.0f;
    for (int n = 0; n < NTOK; ++n) {
        float E = __expf((col[(size_t)n * OUT_DIM] - mx) * INV_TT + EXP_MAX);
        t += rb[n] * E;
    }
    const int idx = b * OUT_DIM + p;
    float cv = c[idx];
    c[idx] = cv / (cv * t + EPS_SK);
}

// ================= kernel 7: SK row pass (optional assignment write) =================
__global__ void k_sk_row(const float* __restrict__ logits,
                         const float* __restrict__ colmax,
                         const float* __restrict__ c,
                         float* __restrict__ r,
                         float* __restrict__ out,
                         int write_out) {
    __shared__ float red[256];
    const int row = blockIdx.x;
    const int b   = row >> 10;
    const size_t base = (size_t)row * OUT_DIM;
    const float* cb = c + b * OUT_DIM;
    const float* mb = colmax + b * OUT_DIM;

    float acc = 0.0f;
#pragma unroll
    for (int s = 0; s < OUT_DIM / 256; ++s) {
        const int p = threadIdx.x + s * 256;
        float E = __expf((logits[base + p] - mb[p]) * INV_TT + EXP_MAX);
        acc += E * cb[p];
    }
    float v = block_sum(acc, red);
    float rv = r[row];
    float rn = rv / (rv * v + EPS_SK);
    if (threadIdx.x == 0) r[row] = rn;

    if (write_out) {
#pragma unroll
        for (int s = 0; s < OUT_DIM / 256; ++s) {
            const int p = threadIdx.x + s * 256;
            float E = __expf((logits[base + p] - mb[p]) * INV_TT + EXP_MAX);
            out[base + p] = rn * E * cb[p];
        }
    }
}

// ================= kernel 8: per-row CE loss =================
__global__ void k_loss_row(const float* __restrict__ logits,
                           const float* __restrict__ tgt,
                           float* __restrict__ rowloss) {
    __shared__ float red[256];
    const int row = blockIdx.x;
    const size_t base = (size_t)row * OUT_DIM;

    float mx = -3.402823466e+38f;
#pragma unroll
    for (int s = 0; s < OUT_DIM / 256; ++s)
        mx = fmaxf(mx, logits[base + threadIdx.x + s * 256] * INV_PT);
    mx = block_max(mx, red);

    float se = 0.0f;
#pragma unroll
    for (int s = 0; s < OUT_DIM / 256; ++s)
        se += __expf(logits[base + threadIdx.x + s * 256] * INV_PT - mx);
    se = block_sum(se, red);
    const float lse = mx + __logf(se);

    float dot = 0.0f;
#pragma unroll
    for (int s = 0; s < OUT_DIM / 256; ++s) {
        const int p = threadIdx.x + s * 256;
        dot += tgt[base + p] * (logits[base + p] * INV_PT - lse);
    }
    dot = block_sum(dot, red);
    if (threadIdx.x == 0) rowloss[row] = -dot;
}

// ================= kernel 9: deterministic final loss reduction =================
__global__ void k_loss_final(const float* __restrict__ rowloss, float* __restrict__ loss_out) {
    __shared__ float red[256];
    float s = 0.0f;
    for (int k = 0; k < NROWS / 256; ++k) s += rowloss[threadIdx.x + k * 256];
    s = block_sum(s, red);
    if (threadIdx.x == 0) loss_out[0] = s / (float)NROWS;
}

// ================= host launcher =================
extern "C" void kernel_launch(void* const* d_in, const int* in_sizes, int n_in,
                              void* d_out, int out_size, void* d_ws, size_t ws_size,
                              hipStream_t stream) {
    const float* x = (const float*)d_in[0];   // [32,1024,768] f32
    const float* W = (const float*)d_in[1];   // [4096,768]    f32
    float* out = (float*)d_out;               // assignments [32768*4096] + loss [1]

    char* ws = (char*)d_ws;
    size_t off = 0;
    float*    logits = (float*)(ws + off);    off += (size_t)NROWS * OUT_DIM * sizeof(float);   // 512 MB
    uint16_t* xb     = (uint16_t*)(ws + off); off += (size_t)NROWS * IN_DIM * sizeof(uint16_t); // 48 MB
    uint16_t* wb     = (uint16_t*)(ws + off); off += (size_t)OUT_DIM * IN_DIM * sizeof(uint16_t);
    float*    colmax = (float*)(ws + off);    off += (size_t)NBATCH * OUT_DIM * sizeof(float);
    float*    cvec   = (float*)(ws + off);    off += (size_t)NBATCH * OUT_DIM * sizeof(float);
    float*    rvec   = (float*)(ws + off);    off += (size_t)NROWS * sizeof(float);
    float*    rowloss= (float*)(ws + off);    off += (size_t)NROWS * sizeof(float);
    (void)ws_size; (void)in_sizes; (void)n_in; (void)out_size;

    k_rownorm_cvt<<<NROWS, 256, 0, stream>>>(x, xb);
    k_cvt_w<<<(OUT_DIM * IN_DIM + 255) / 256, 256, 0, stream>>>(W, wb);

    {
        dim3 grid(OUT_DIM / 256, NROWS / 32);
        k_gemm_wmma<<<grid, 128, 0, stream>>>(xb, wb, logits);
    }

    k_init_rc<<<(NBATCH * OUT_DIM + 255) / 256, 256, 0, stream>>>(rvec, cvec);
    {
        dim3 grid(OUT_DIM / 256, NBATCH);
        k_colmax<<<grid, 256, 0, stream>>>(logits, colmax);
    }

    for (int it = 0; it < 3; ++it) {
        dim3 cgrid(OUT_DIM / 256, NBATCH);
        k_sk_col<<<cgrid, 256, 0, stream>>>(logits, colmax, rvec, cvec);
        k_sk_row<<<NROWS, 256, 0, stream>>>(logits, colmax, cvec, rvec, out,
                                            it == 2 ? 1 : 0);
    }

    k_loss_row<<<NROWS, 256, 0, stream>>>(logits, out, rowloss);
    k_loss_final<<<1, 256, 0, stream>>>(rowloss, out + (size_t)NROWS * OUT_DIM);
}
